// HGC_lstm_85727547228594
// MI455X (gfx1250) — matrix operand order repaired and checked by reading, not compiled, not run on hardware
//
#include <hip/hip_runtime.h>
#include <hip/hip_bf16.h>

// ---------------------------------------------------------------------------
// MI455X (gfx1250) implementation. wave32, WMMA bf16 (f32 accumulate) for all
// dense GEMMs and the LSTM recurrence; fp32 atomics for the sparse channels.
// ---------------------------------------------------------------------------

typedef __attribute__((ext_vector_type(16))) __bf16 v16bf;
typedef __attribute__((ext_vector_type(8)))  float  v8f;

union FragBF { v16bf v; unsigned u[8]; };

#define DIMF 128

__device__ __forceinline__ unsigned short f2bf(float f) {
  union { float f; unsigned u; } c; c.f = f;
  unsigned u = c.u;
  u += 0x7FFFu + ((u >> 16) & 1u);          // round-to-nearest-even
  return (unsigned short)(u >> 16);
}
__device__ __forceinline__ float sigf(float x) { return 1.f / (1.f + __expf(-x)); }
__device__ __forceinline__ void atomAddF(float* p, float v) { unsafeAtomicAdd(p, v); }

// ---------------------------------------------------------------------------
// Generic bf16 WMMA GEMM:  Out(MxN) = epilogue( A(MxK,f32) * B + bias )
// B is f32, either KxN (bTrans=0) or NxK (bTrans=1). Block tile 64x32, 8 waves,
// each wave owns one 16x16 WMMA tile chain over K (step 32).
// epi: 0 = plain (+biases), 1 = relu(+biases), 2 = gate: A*sigmoid(acc+bias)
// ---------------------------------------------------------------------------
#define GBM 64
#define GBN 32
#define GBK 32

__global__ __launch_bounds__(256) void gemm_bf16_kernel(
    const float* __restrict__ A, const float* __restrict__ Bm,
    const float* __restrict__ bias1, const float* __restrict__ bias2,
    float* __restrict__ Out, int M, int N, int K, int bTrans, int epi)
{
  __shared__ unsigned short Asm[GBM][GBK];   // [m][k]
  __shared__ unsigned short Bsm[GBN][GBK];   // [n][k]  (k pairs contiguous)
  const int tid  = threadIdx.x;
  const int lane = tid & 31;
  const int wave = tid >> 5;                 // 0..7
  const int wm   = wave >> 1;                // 0..3
  const int wn   = wave & 1;                 // 0..1
  const int mbase = blockIdx.y * GBM;
  const int nbase = blockIdx.x * GBN;
  const int half = lane >> 4;
  const int l16  = lane & 15;

  v8f acc;
#pragma unroll
  for (int v = 0; v < 8; ++v) acc[v] = 0.f;

  for (int k0 = 0; k0 < K; k0 += GBK) {
    // stage A tile (64x32) as bf16
    for (int i = tid; i < GBM * GBK; i += 256) {
      int m = i >> 5, k = i & 31;
      int gr = mbase + m;
      float v = (gr < M) ? A[(size_t)gr * K + k0 + k] : 0.f;
      Asm[m][k] = f2bf(v);
    }
    // stage B tile (32n x 32k) as bf16, n-major so k pairs are 4B-aligned
    for (int i = tid; i < GBN * GBK; i += 256) {
      int n = i >> 5, k = i & 31;
      float v = bTrans ? Bm[(size_t)(nbase + n) * K + k0 + k]
                       : Bm[(size_t)(k0 + k) * N + nbase + n];
      Bsm[n][k] = f2bf(v);
    }
    if (k0 + GBK < K && tid < GBM) {
      int gr = mbase + tid; if (gr >= M) gr = M - 1;
      __builtin_prefetch(&A[(size_t)gr * K + k0 + GBK], 0, 1);
    }
    __syncthreads();

    FragBF af, bf;
    const int am = wm * 16 + l16;
#pragma unroll
    for (int v = 0; v < 8; ++v) {
      int kk = 16 * half + 2 * v;
      af.u[v] = *(const unsigned*)&Asm[am][kk];
    }
    const int bn = wn * 16 + l16;
#pragma unroll
    for (int v = 0; v < 8; ++v) {
      int kk = 16 * half + 2 * v;
      bf.u[v] = *(const unsigned*)&Bsm[bn][kk];
    }
    acc = __builtin_amdgcn_wmma_f32_16x16x32_bf16(false, af.v, false, bf.v,
                                                  (short)0, acc, false, false);
    __syncthreads();
  }

  const int col = nbase + wn * 16 + l16;
#pragma unroll
  for (int v = 0; v < 8; ++v) {
    int row = mbase + wm * 16 + v + 8 * half;
    if (row < M) {
      float z = acc[v];
      if (bias1) z += bias1[col];
      if (bias2) z += bias2[col];
      if (epi == 1) z = fmaxf(z, 0.f);
      else if (epi == 2) z = A[(size_t)row * K + col] * sigf(z);  // K==N==128
      Out[(size_t)row * N + col] = z;
    }
  }
}

// ---------------------------------------------------------------------------
// SpMM edge scatter: Y[rows[e]] += vals[e] * X[cols[e]]   (one wave per edge)
// ---------------------------------------------------------------------------
__global__ void spmm_edges_kernel(const int* __restrict__ rows, const int* __restrict__ cols,
                                  const float* __restrict__ vals, const float* __restrict__ X,
                                  float* __restrict__ Y, int nnz)
{
  int e = blockIdx.x * (blockDim.x >> 5) + (threadIdx.x >> 5);
  int lane = threadIdx.x & 31;
  if (e >= nnz) return;
  int r = rows[e], c = cols[e];
  float v = vals[e];
  float4 x4 = ((const float4*)(X + (size_t)c * DIMF))[lane];
  float* yp = Y + (size_t)r * DIMF + lane * 4;
  atomAddF(yp + 0, v * x4.x); atomAddF(yp + 1, v * x4.y);
  atomAddF(yp + 2, v * x4.z); atomAddF(yp + 3, v * x4.w);
}

// GCN edge scatter: Y[dst] += dinv[src]*dinv[dst]*H[src]
__global__ void gcn_edge_kernel(const int* __restrict__ src, const int* __restrict__ dst,
                                const float* __restrict__ dinv, const float* __restrict__ H,
                                float* __restrict__ Y, int nnz)
{
  int e = blockIdx.x * (blockDim.x >> 5) + (threadIdx.x >> 5);
  int lane = threadIdx.x & 31;
  if (e >= nnz) return;
  int s = src[e], d = dst[e];
  float nv = dinv[s] * dinv[d];
  float4 hv = ((const float4*)(H + (size_t)s * DIMF))[lane];
  float* yp = Y + (size_t)d * DIMF + lane * 4;
  atomAddF(yp + 0, nv * hv.x); atomAddF(yp + 1, nv * hv.y);
  atomAddF(yp + 2, nv * hv.z); atomAddF(yp + 3, nv * hv.w);
}

// ---------------------------------------------------------------------------
// Gather + gated-channel fusion:
//   row = (x1[idx]+x2[idx]+x3[idx])*scale ; out (+)= sigmoid(row.w + b)*row
// ---------------------------------------------------------------------------
__global__ void fuse_gather_kernel(const int* __restrict__ idx,
    const float* __restrict__ x1, const float* __restrict__ x2, const float* __restrict__ x3,
    float scale, const float* __restrict__ w, const float* __restrict__ bsc,
    float* __restrict__ out, int n, int accumulate)
{
  int i = blockIdx.x * (blockDim.x >> 5) + (threadIdx.x >> 5);
  int lane = threadIdx.x & 31;
  if (i >= n) return;
  size_t g = (size_t)idx[i] * DIMF;
  float4 r = ((const float4*)(x1 + g))[lane];
  if (x2) { float4 t = ((const float4*)(x2 + g))[lane]; r.x += t.x; r.y += t.y; r.z += t.z; r.w += t.w; }
  if (x3) { float4 t = ((const float4*)(x3 + g))[lane]; r.x += t.x; r.y += t.y; r.z += t.z; r.w += t.w; }
  r.x *= scale; r.y *= scale; r.z *= scale; r.w *= scale;
  float4 wv = ((const float4*)w)[lane];
  float p = r.x * wv.x + r.y * wv.y + r.z * wv.z + r.w * wv.w;
  for (int o = 16; o; o >>= 1) p += __shfl_down(p, o, 32);
  p = __shfl(p, 0, 32);
  float gt = sigf(p + bsc[0]);
  float* op = out + (size_t)i * DIMF + lane * 4;
  if (accumulate) { op[0] += gt * r.x; op[1] += gt * r.y; op[2] += gt * r.z; op[3] += gt * r.w; }
  else            { *(float4*)op = make_float4(gt * r.x, gt * r.y, gt * r.z, gt * r.w); }
}

// ---------------------------------------------------------------------------
// Batched LSTM recurrence. 8 blocks x 16 batch rows; 8 waves per block, wave w
// owns hidden chunk hj = 16w..16w+15 across all four gates. w_hh fragments are
// loaded to VGPRs once; c lives in accumulator registers; h round-trips via
// 4KB LDS (bf16). xpre already contains x@w_ih.T + b_ih + b_hh.
// ---------------------------------------------------------------------------
__global__ __launch_bounds__(256) void lstm_kernel(
    const float* __restrict__ xpre,   // [B*T, 512]  n = b*T+t
    const float* __restrict__ w_hh,   // [512, 128]
    float* __restrict__ hs,           // [B*T, 128]
    int T_, int B_)
{
  __shared__ unsigned short hbf[16 * DIMF];
  const int tid  = threadIdx.x;
  const int lane = tid & 31;
  const int wave = tid >> 5;
  const int half = lane >> 4, l16 = lane & 15;
  const int bbase = blockIdx.x * 16;

  FragBF Bf[4][4];                    // [gate][kchunk], constant over t
#pragma unroll
  for (int g = 0; g < 4; ++g) {
    int j = g * DIMF + wave * 16 + l16;      // row of w_hh (gate output col)
#pragma unroll
    for (int kc = 0; kc < 4; ++kc) {
#pragma unroll
      for (int v = 0; v < 8; ++v) {
        int k = kc * 32 + 16 * half + 2 * v;
        unsigned lo = f2bf(w_hh[(size_t)j * DIMF + k]);
        unsigned hi = f2bf(w_hh[(size_t)j * DIMF + k + 1]);
        Bf[g][kc].u[v] = lo | (hi << 16);
      }
    }
  }
  for (int i = tid; i < 16 * DIMF; i += 256) hbf[i] = 0;   // h0 = 0
  float cst[8];
#pragma unroll
  for (int v = 0; v < 8; ++v) cst[v] = 0.f;                // c0 = 0
  __syncthreads();

  for (int t = 0; t < T_; ++t) {
    FragBF Af[4];
#pragma unroll
    for (int kc = 0; kc < 4; ++kc)
#pragma unroll
      for (int v = 0; v < 8; ++v) {
        int k = kc * 32 + 16 * half + 2 * v;
        Af[kc].u[v] = *(const unsigned*)&hbf[l16 * DIMF + k];
      }

    v8f z[4];
#pragma unroll
    for (int g = 0; g < 4; ++g) {
      v8f acc;
#pragma unroll
      for (int v = 0; v < 8; ++v) {
        int m = v + 8 * half;
        acc[v] = xpre[((size_t)(bbase + m) * T_ + t) * 512 + g * DIMF + wave * 16 + l16];
      }
#pragma unroll
      for (int kc = 0; kc < 4; ++kc)
        acc = __builtin_amdgcn_wmma_f32_16x16x32_bf16(false, Af[kc].v, false, Bf[g][kc].v,
                                                      (short)0, acc, false, false);
      z[g] = acc;
    }
    __syncthreads();   // all hbf reads done before overwrite

#pragma unroll
    for (int v = 0; v < 8; ++v) {
      float ig = sigf(z[0][v]);
      float fg = sigf(z[1][v]);
      float gg = tanhf(z[2][v]);
      float og = sigf(z[3][v]);
      float cn = fg * cst[v] + ig * gg;
      cst[v] = cn;
      float hn = og * tanhf(cn);
      int m = v + 8 * half;
      int hj = wave * 16 + l16;
      hbf[m * DIMF + hj] = f2bf(hn);
      hs[((size_t)(bbase + m) * T_ + t) * DIMF + hj] = hn;
    }
    __syncthreads();   // new h visible to all waves
  }
}

// ---------------------------------------------------------------------------
// small elementwise helpers (grid-stride)
// ---------------------------------------------------------------------------
__global__ void ew_fill(float* p, float v, int n) {
  for (int i = blockIdx.x * blockDim.x + threadIdx.x; i < n; i += gridDim.x * blockDim.x) p[i] = v;
}
__global__ void ew_copy(const float* __restrict__ a, float* __restrict__ o, int n) {
  for (int i = blockIdx.x * blockDim.x + threadIdx.x; i < n; i += gridDim.x * blockDim.x) o[i] = a[i];
}
__global__ void ew_mean3(const float* __restrict__ a, const float* __restrict__ b,
                         const float* __restrict__ c, float* __restrict__ o, int n) {
  for (int i = blockIdx.x * blockDim.x + threadIdx.x; i < n; i += gridDim.x * blockDim.x)
    o[i] = (a[i] + b[i] + c[i]) * (1.f / 3.f);
}
__global__ void count_deg_kernel(const int* __restrict__ dst, float* deg, int nnz) {
  for (int i = blockIdx.x * blockDim.x + threadIdx.x; i < nnz; i += gridDim.x * blockDim.x)
    atomAddF(&deg[dst[i]], 1.f);
}
__global__ void dinv_kernel(float* deg, int n) {
  for (int i = blockIdx.x * blockDim.x + threadIdx.x; i < n; i += gridDim.x * blockDim.x)
    deg[i] = rsqrtf(fmaxf(deg[i], 1.f));
}
__global__ void gcn_self_kernel(const float* __restrict__ h, const float* __restrict__ dinv,
                                float* __restrict__ y, int n) {      // n = rows*128
  for (int i = blockIdx.x * blockDim.x + threadIdx.x; i < n; i += gridDim.x * blockDim.x) {
    int r = i >> 7;
    y[i] = dinv[r] * dinv[r] * h[i];
  }
}
__global__ void bias_relu_kernel(const float* __restrict__ y, const float* __restrict__ b,
                                 float* __restrict__ o, int n) {     // n = rows*128
  for (int i = blockIdx.x * blockDim.x + threadIdx.x; i < n; i += gridDim.x * blockDim.x)
    o[i] = fmaxf(y[i] + b[i & 127], 0.f);
}
__global__ void build_lstm_in_kernel(const float* __restrict__ fq, const float* __restrict__ fs,
                                     const float* __restrict__ ce, const int* __restrict__ ans,
                                     float* __restrict__ o, int n) { // n = BT*320
  for (int i = blockIdx.x * blockDim.x + threadIdx.x; i < n; i += gridDim.x * blockDim.x) {
    int row = i / 320, j = i - row * 320;
    o[i] = (j < 128) ? fq[(size_t)row * 128 + j]
         : (j < 256) ? fs[(size_t)row * 128 + (j - 128)]
                     : ce[ans[row] * 64 + (j - 256)];
  }
}
__global__ void build_pred_in_kernel(const float* __restrict__ hs, const float* __restrict__ fq,
                                     const float* __restrict__ fs, float* __restrict__ o, int n) { // BT*384
  for (int i = blockIdx.x * blockDim.x + threadIdx.x; i < n; i += gridDim.x * blockDim.x) {
    int row = i / 384, j = i - row * 384;
    o[i] = (j < 128) ? hs[(size_t)row * 128 + j]
         : (j < 256) ? fq[(size_t)row * 128 + (j - 128)]
                     : fs[(size_t)row * 128 + (j - 256)];
  }
}
__global__ void final_pred_kernel(const float* __restrict__ hid, const float* __restrict__ w,
                                  const float* __restrict__ b, float* __restrict__ out, int n) {
  int i = blockIdx.x * (blockDim.x >> 5) + (threadIdx.x >> 5);
  int lane = threadIdx.x & 31;
  if (i >= n) return;
  float4 hv = ((const float4*)(hid + (size_t)i * DIMF))[lane];
  float4 wv = ((const float4*)w)[lane];
  float p = hv.x * wv.x + hv.y * wv.y + hv.z * wv.z + hv.w * wv.w;
  for (int o = 16; o; o >>= 1) p += __shfl_down(p, o, 32);
  if (lane == 0) out[i] = sigf(p + b[0]);
}

// ---------------------------------------------------------------------------
extern "C" void kernel_launch(void* const* d_in, const int* in_sizes, int n_in,
                              void* d_out, int out_size, void* d_ws, size_t ws_size,
                              hipStream_t stream) {
  (void)in_sizes; (void)n_in; (void)out_size; (void)ws_size;
  const int NQ = 100000, NS = 2000, T = 500, Bb = 128, BT = Bb * T;
  const int NNZ_HG = 300000, NNZ_GCN = 64000, NNZ_SIM = 1000000;

  const int*   question_seq = (const int*)d_in[0];
  const int*   skill_seq    = (const int*)d_in[1];
  const int*   answer_seq   = (const int*)d_in[2];
  const int*   hg_q   = (const int*)d_in[3];
  const int*   hg_c   = (const int*)d_in[4];
  const int*   gcn_src = (const int*)d_in[5];
  const int*   gcn_dst = (const int*)d_in[6];
  const int*   sim_src = (const int*)d_in[7];
  const int*   sim_dst = (const int*)d_in[8];
  const float* hg_vals_qc = (const float*)d_in[9];
  const float* hg_vals_cq = (const float*)d_in[10];
  const float* sim_vals   = (const float*)d_in[11];
  const float* q_emb = (const float*)d_in[12];
  const float* s_emb = (const float*)d_in[13];
  const float* c_emb = (const float*)d_in[14];
  const float* w_g1 = (const float*)d_in[15];
  const float* w_g2 = (const float*)d_in[16];
  const float* w_g3 = (const float*)d_in[17];
  const float* b_g1 = (const float*)d_in[18];
  const float* b_g2 = (const float*)d_in[19];
  const float* b_g3 = (const float*)d_in[20];
  const float* gcn_w1 = (const float*)d_in[21];
  const float* gcn_b1 = (const float*)d_in[22];
  const float* gcn_w2 = (const float*)d_in[23];
  const float* gcn_b2 = (const float*)d_in[24];
  const float* gcn_w3 = (const float*)d_in[25];
  const float* gcn_b3 = (const float*)d_in[26];
  const float* w_sg1 = (const float*)d_in[27];
  const float* w_sg2 = (const float*)d_in[28];
  const float* w_qg1 = (const float*)d_in[29];
  const float* w_qg3 = (const float*)d_in[30];
  const float* b_sg1 = (const float*)d_in[31];
  const float* b_sg2 = (const float*)d_in[32];
  const float* b_qg1 = (const float*)d_in[33];
  const float* b_qg3 = (const float*)d_in[34];
  const float* w_ih = (const float*)d_in[35];
  const float* w_hh = (const float*)d_in[36];
  const float* b_ih = (const float*)d_in[37];
  const float* b_hh = (const float*)d_in[38];
  const float* fc1_w = (const float*)d_in[39];
  const float* fc1_b = (const float*)d_in[40];
  const float* fc2_w = (const float*)d_in[41];
  const float* fc2_b = (const float*)d_in[42];
  float* outp = (float*)d_out;

  // ---- workspace layout ----
  size_t off = 0;
  auto alloc = [&](size_t nFloats) -> float* {
    float* p = (float*)((char*)d_ws + off);
    off += ((nFloats * sizeof(float) + 255) & ~(size_t)255);
    return p;
  };
  const size_t NQD = (size_t)NQ * 128, NSD = (size_t)NS * 128;
  float* Qbase = alloc(3 * NQD);
  float* QA = Qbase;                 // three_gate (ch3 e0), later reused
  float* QB = Qbase + NQD;           // ch3 e2 / ch1 one_q
  float* QC = Qbase + 2 * NQD;       // ch3 e1 / ch1 m-temp
  float* SA = alloc(NSD);            // one_gate (ch1 e0)
  float* SB = alloc(NSD);            // two_gate
  float* SC = alloc(NSD);            // ch1 e1
  float* SE = alloc(NSD);            // ch1 e2
  float* SD = alloc(NSD);            // one_skill
  float* Gh = alloc(NSD);
  float* Gy = alloc(NSD);
  float* Gx = alloc(NSD);            // GCN activations / two_skill
  float* deg = alloc(NS);
  float* fq = alloc((size_t)BT * 128);
  float* fs = alloc((size_t)BT * 128);
  float* Pbase = alloc((size_t)BT * 384);   // lstm_in (BT*320) then pred_in (BT*384)
  float* hsb = alloc((size_t)BT * 128);
  float* hid = alloc((size_t)BT * 128);
  float* xpre = Qbase;               // BT*512 = 32.8M floats <= 38.4M; Q dead by then

  auto gemm = [&](const float* A, const float* Bm, const float* b1, const float* b2,
                  float* O, int M, int N, int K, int tr, int epi) {
    dim3 g((N + GBN - 1) / GBN, (M + GBM - 1) / GBM);
    gemm_bf16_kernel<<<g, 256, 0, stream>>>(A, Bm, b1, b2, O, M, N, K, tr, epi);
  };
  auto ewg = [](long n) { long b = (n + 1023) / 1024; return (int)(b > 65535 ? 65535 : b); };
  auto spmm = [&](const int* r, const int* c, const float* v, const float* X, float* Y, int nnz) {
    spmm_edges_kernel<<<(nnz + 7) / 8, 256, 0, stream>>>(r, c, v, X, Y, nnz);
  };

  // ---- gated embeddings (WMMA + fused sigmoid-gate epilogue) ----
  gemm(s_emb, w_g1, b_g1, nullptr, SA, NS, 128, 128, 0, 2);
  gemm(s_emb, w_g2, b_g2, nullptr, SB, NS, 128, 128, 0, 2);
  gemm(q_emb, w_g3, b_g3, nullptr, QA, NQ, 128, 128, 0, 2);

  // ---- channel 3: similarity conv, residual, mean over layers ----
  ew_copy<<<ewg(NQD), 256, 0, stream>>>(QA, QC, (int)NQD);            // init = residual
  spmm(sim_src, sim_dst, sim_vals, QA, QC, NNZ_SIM);                  // e1
  ew_copy<<<ewg(NQD), 256, 0, stream>>>(QC, QB, (int)NQD);
  spmm(sim_src, sim_dst, sim_vals, QC, QB, NNZ_SIM);                  // e2
  // fused_q  = sig(three_q.w_qg3+b)*three_q ; three_q = mean(QA,QC,QB)
  fuse_gather_kernel<<<(BT + 7) / 8, 256, 0, stream>>>(question_seq, QA, QB, QC,
      1.f / 3.f, w_qg3, b_qg3, fq, BT, 0);

  // ---- channel 1: hypergraph conv ----
  ew_fill<<<ewg(NQD), 256, 0, stream>>>(QC, 0.f, (int)NQD);
  spmm(hg_q, hg_c, hg_vals_qc, SA, QC, NNZ_HG);                       // m1
  ew_copy<<<ewg(NSD), 256, 0, stream>>>(SA, SC, (int)NSD);
  spmm(hg_c, hg_q, hg_vals_cq, QC, SC, NNZ_HG);                       // e1
  ew_fill<<<ewg(NQD), 256, 0, stream>>>(QC, 0.f, (int)NQD);
  spmm(hg_q, hg_c, hg_vals_qc, SC, QC, NNZ_HG);                       // m2
  ew_copy<<<ewg(NSD), 256, 0, stream>>>(SC, SE, (int)NSD);
  spmm(hg_c, hg_q, hg_vals_cq, QC, SE, NNZ_HG);                       // e2
  ew_mean3<<<ewg(NSD), 256, 0, stream>>>(SA, SC, SE, SD, (int)NSD);   // one_skill
  ew_fill<<<ewg(NQD), 256, 0, stream>>>(QB, 0.f, (int)NQD);
  spmm(hg_q, hg_c, hg_vals_qc, SD, QB, NNZ_HG);                       // one_q
  fuse_gather_kernel<<<(BT + 7) / 8, 256, 0, stream>>>(question_seq, QB, nullptr, nullptr,
      1.f, w_qg1, b_qg1, fq, BT, 1);                                  // fused_q +=

  // ---- channel 2: 3-layer GCN ----
  ew_fill<<<ewg(NS), 256, 0, stream>>>(deg, 1.f, NS);                 // self loops
  count_deg_kernel<<<ewg(NNZ_GCN), 256, 0, stream>>>(gcn_dst, deg, NNZ_GCN);
  dinv_kernel<<<ewg(NS), 256, 0, stream>>>(deg, NS);
  const float* gw[3] = {gcn_w1, gcn_w2, gcn_w3};
  const float* gb[3] = {gcn_b1, gcn_b2, gcn_b3};
  const float* gin = SB;
  for (int l = 0; l < 3; ++l) {
    gemm(gin, gw[l], nullptr, nullptr, Gh, NS, 128, 128, 0, 0);       // h = x@W
    gcn_self_kernel<<<ewg(NSD), 256, 0, stream>>>(Gh, deg, Gy, (int)NSD);
    gcn_edge_kernel<<<(NNZ_GCN + 7) / 8, 256, 0, stream>>>(gcn_src, gcn_dst, deg, Gh, Gy, NNZ_GCN);
    bias_relu_kernel<<<ewg(NSD), 256, 0, stream>>>(Gy, gb[l], Gx, (int)NSD);
    gin = Gx;
  }

  // ---- fused skill channel ----
  fuse_gather_kernel<<<(BT + 7) / 8, 256, 0, stream>>>(skill_seq, SD, nullptr, nullptr,
      1.f, w_sg1, b_sg1, fs, BT, 0);
  fuse_gather_kernel<<<(BT + 7) / 8, 256, 0, stream>>>(skill_seq, Gx, nullptr, nullptr,
      1.f, w_sg2, b_sg2, fs, BT, 1);

  // ---- LSTM: precompute input projection (WMMA), then batched recurrence ----
  build_lstm_in_kernel<<<ewg((long)BT * 320), 256, 0, stream>>>(fq, fs, c_emb, answer_seq,
      Pbase, BT * 320);
  gemm(Pbase, w_ih, b_ih, b_hh, xpre, BT, 512, 320, 1, 0);            // w_ih is (N,K)
  lstm_kernel<<<Bb / 16, 256, 0, stream>>>(xpre, w_hh, hsb, T, Bb);

  // ---- prediction head ----
  build_pred_in_kernel<<<ewg((long)BT * 384), 256, 0, stream>>>(hsb, fq, fs, Pbase, BT * 384);
  gemm(Pbase, fc1_w, fc1_b, nullptr, hid, BT, 128, 384, 0, 1);        // relu
  final_pred_kernel<<<(BT + 7) / 8, 256, 0, stream>>>(hid, fc2_w, fc2_b, outp, BT);
}
